// GAT_55035710931201
// MI455X (gfx1250) — matrix operand order repaired and checked by reading, not compiled, not run on hardware
//
#include <hip/hip_runtime.h>
#include <hip/hip_bf16.h>

// ---------------- problem constants (match reference) ----------------
#define NN    50000
#define EE    800000
#define FIN   128
#define EDIM  16
#define HIDC  32
#define NHEAD 4
#define HC1   128          // NHEAD*HIDC
#define GG    64
#define COUT  2
#define ETOT  (EE + NN)    // edges + self loops
#define NEG_SLOPE 0.2f

typedef __attribute__((ext_vector_type(16))) _Float16 v16h;
typedef __attribute__((ext_vector_type(8)))  _Float16 v8h;
typedef __attribute__((ext_vector_type(8)))  float    v8f;

// -------- monotonic float<->uint encoding for atomicMax on floats -----
__device__ __forceinline__ unsigned fenc(float f) {
  unsigned u = __float_as_uint(f);
  return (u & 0x80000000u) ? ~u : (u | 0x80000000u);
}
__device__ __forceinline__ float fdec(unsigned u) {
  unsigned v = (u & 0x80000000u) ? (u & 0x7fffffffu) : ~u;
  return __uint_as_float(v);
}
#define FENC_NEG_INF 0x007FFFFFu   // fenc(-inf)

// ---------------------------- utility fills ---------------------------
__global__ void fill_u32_kernel(unsigned* p, unsigned v, size_t n) {
  size_t i = (size_t)blockIdx.x * blockDim.x + threadIdx.x;
  if (i < n) p[i] = v;
}

// ---------------- f32 -> f16 elementwise conversion -------------------
__global__ void cvt_f16_kernel(const float* __restrict__ x,
                               _Float16* __restrict__ y, size_t n) {
  size_t i = (size_t)blockIdx.x * blockDim.x + threadIdx.x;
  if (i < n) y[i] = (_Float16)x[i];
}

// ------- pack W[K,NOUT] f32 into per-lane B-fragment order, f16 -------
// Wp[((t*(K/32)+ks)*32 + lane)*16 + i] = W[(ks*32 + 16*(lane>>4) + i)*NOUT
//                                          + t*16 + (lane&15)]
// so each lane's 16 B halves for (tile t, K-step ks) are 32 contiguous B.
template<int K, int NOUT>
__global__ void pack_w_kernel(const float* __restrict__ W,
                              _Float16* __restrict__ Wp) {
  int idx = blockIdx.x * blockDim.x + threadIdx.x;
  if (idx >= K * NOUT) return;
  int i    = idx & 15;
  int lane = (idx >> 4) & 31;
  int ks   = (idx >> 9) % (K / 32);
  int t    = idx / (16 * 32 * (K / 32));
  int k = ks * 32 + 16 * (lane >> 4) + i;
  int n = t * 16 + (lane & 15);
  Wp[idx] = (_Float16)W[(size_t)k * NOUT + n];
}

// ------------------- WMMA GEMM: Y[M,NOUT] = Xh[M,K] @ W[K,NOUT] -------
// Xh is f16 row-major; Wp is the pre-packed f16 fragment layout above.
// 4 waves per 128-thread block; each wave owns rows [16*tile, 16*tile+16)
// and computes all NOUT/16 column tiles, so the A fragment (two b128
// loads per K-step) is reused across tiles. M = 50000 = 3125*16.
template<int K, int NOUT>
__global__ __launch_bounds__(128) void gemm_wmma_kernel(
    const _Float16* __restrict__ Xh, const _Float16* __restrict__ Wp,
    float* __restrict__ Y) {
  constexpr int NT = NOUT / 16;
  constexpr int NTILE = NN / 16;
  const int lane = threadIdx.x & 31;
  const int wid  = threadIdx.x >> 5;
  const int tile = blockIdx.x * 4 + wid;          // wave-uniform
  if (tile >= NTILE) return;
  const int half = lane >> 4;
  const int l16  = lane & 15;
  const int row0 = tile * 16;

  v8f acc[NT] = {};
  for (int kb = 0; kb < K; kb += 32) {
    // A fragment: halves i -> K=kb+16*half+i
    const v8h* xa = (const v8h*)(Xh + (size_t)(row0 + l16) * K + kb + 16 * half);
    v8h alo = xa[0];
    v8h ahi = xa[1];                               // +8 elements
    v16h a = __builtin_shufflevector(alo, ahi,
        0, 1, 2, 3, 4, 5, 6, 7, 8, 9, 10, 11, 12, 13, 14, 15);
#pragma unroll
    for (int t = 0; t < NT; ++t) {
      const v8h* wb = (const v8h*)(Wp +
          ((size_t)(t * (K / 32) + (kb >> 5)) * 32 + lane) * 16);
      v16h b = __builtin_shufflevector(wb[0], wb[1],
          0, 1, 2, 3, 4, 5, 6, 7, 8, 9, 10, 11, 12, 13, 14, 15);
      acc[t] = __builtin_amdgcn_wmma_f32_16x16x32_f16(
          false, a, false, b, (short)0, acc[t], false, false);
    }
  }
  // D: lane = col l16, VGPR r -> row = r + 8*half
#pragma unroll
  for (int t = 0; t < NT; ++t) {
    float* yout = Y + (size_t)t * 16 + l16;
#pragma unroll
    for (int r = 0; r < 8; ++r)
      yout[(size_t)(row0 + r + 8 * half) * NOUT] = acc[t][r];
  }
}

// ------------- self-loop attr: degree + mean of incoming attrs --------
__global__ void degree_loop_accum_kernel(const int* __restrict__ dst,
                                         const float* __restrict__ eattr,
                                         float* __restrict__ cnt,
                                         float* __restrict__ loop_attr) {
  int e = blockIdx.x * blockDim.x + threadIdx.x;
  if (e >= EE) return;
  int d = dst[e];
  atomicAdd(&cnt[d], 1.0f);
  const float* ea = eattr + (size_t)e * EDIM;
#pragma unroll
  for (int k = 0; k < EDIM; ++k)
    atomicAdd(&loop_attr[(size_t)d * EDIM + k], ea[k]);
}

__global__ void loop_div_kernel(float* __restrict__ loop_attr,
                                const float* __restrict__ cnt) {
  size_t i = (size_t)blockIdx.x * blockDim.x + threadIdx.x;
  if (i >= (size_t)NN * EDIM) return;
  loop_attr[i] /= fmaxf(cnt[i / EDIM], 1.0f);
}

// --------- fused per-(edge,head) logits + running segment max ---------
template<int H, int CH, int HC>
__global__ void edge_logits_kernel(const int* __restrict__ srcI,
                                   const int* __restrict__ dstI,
                                   const float* __restrict__ eattr,
                                   const float* __restrict__ loop_attr,
                                   const float* __restrict__ xl,
                                   const float* __restrict__ xr,
                                   const float* __restrict__ We,   // [EDIM,HC]
                                   const float* __restrict__ att,  // [H,CH]
                                   float* __restrict__ logits,     // [ETOT,H]
                                   unsigned* __restrict__ mx) {    // [NN,H]
  long t = (long)blockIdx.x * blockDim.x + threadIdx.x;
  int e = (int)(t / H), h = (int)(t % H);
  if (e >= ETOT) return;
  int s, d; const float* ea;
  if (e < EE) { s = srcI[e]; d = dstI[e]; ea = eattr + (size_t)e * EDIM; }
  else        { s = d = e - EE;           ea = loop_attr + (size_t)s * EDIM; }

  float eav[EDIM];
#pragma unroll
  for (int k = 0; k < EDIM; ++k) eav[k] = ea[k];

  const float* xls = xl + (size_t)s * HC + h * CH;
  const float* xrd = xr + (size_t)d * HC + h * CH;
  const float* at  = att + h * CH;
  float logit = 0.0f;
#pragma unroll 4
  for (int c = 0; c < CH; ++c) {
    const float* wc = We + h * CH + c;
    float em = 0.0f;
#pragma unroll
    for (int k = 0; k < EDIM; ++k) em = fmaf(eav[k], wc[(size_t)k * HC], em);
    float m = xls[c] + xrd[c] + em;
    m = (m > 0.0f) ? m : NEG_SLOPE * m;          // leaky relu
    logit = fmaf(m, at[c], logit);
  }
  logits[(size_t)e * H + h] = logit;
  atomicMax(&mx[(size_t)d * H + h], fenc(logit));
}

// ----------------- exp(logit - max) and segment-sum denom -------------
template<int H>
__global__ void edge_expden_kernel(const int* __restrict__ dstI,
                                   const unsigned* __restrict__ mx,
                                   float* __restrict__ logits,   // in: logit, out: a
                                   float* __restrict__ den) {
  long t = (long)blockIdx.x * blockDim.x + threadIdx.x;
  int e = (int)(t / H), h = (int)(t % H);
  if (e >= ETOT) return;
  int d = (e < EE) ? dstI[e] : (e - EE);
  float a = __expf(logits[(size_t)e * H + h] - fdec(mx[(size_t)d * H + h]));
  logits[(size_t)e * H + h] = a;
  atomicAdd(&den[(size_t)d * H + h], a);
}

// ---------------- alpha * xl[src] scatter-add into dst ----------------
template<int H, int CH, int HC>
__global__ void edge_aggregate_kernel(const int* __restrict__ srcI,
                                      const int* __restrict__ dstI,
                                      const float* __restrict__ anum, // [ETOT,H]
                                      const float* __restrict__ den,  // [NN,H]
                                      const float* __restrict__ xl,
                                      float* __restrict__ out) {      // [NN,HC]
  long t = (long)blockIdx.x * blockDim.x + threadIdx.x;
  int e = (int)(t / H), h = (int)(t % H);
  if (e >= ETOT) return;
  int s, d;
  if (e < EE) { s = srcI[e]; d = dstI[e]; } else { s = d = e - EE; }
  float alpha = anum[(size_t)e * H + h] / (den[(size_t)d * H + h] + 1e-16f);
  const float* xls = xl + (size_t)s * HC + h * CH;
  float* o = out + (size_t)d * HC + h * CH;
#pragma unroll 8
  for (int c = 0; c < CH; ++c) atomicAdd(&o[c], alpha * xls[c]);
}

// ---------- bias + relu, emitting the f16 copy for layer-2 GEMM -------
__global__ void bias_relu_f16_kernel(const float* __restrict__ h,
                                     const float* __restrict__ b,
                                     _Float16* __restrict__ out16,
                                     int cols, size_t total) {
  size_t i = (size_t)blockIdx.x * blockDim.x + threadIdx.x;
  if (i >= total) return;
  float v = h[i] + b[i % cols];
  out16[i] = (_Float16)((v > 0.0f) ? v : 0.0f);
}

// ----------------- fused layer2 bias+relu + mean pool -----------------
__global__ void pool_kernel(const float* __restrict__ h2, const int* __restrict__ batch,
                            const float* __restrict__ b2,
                            float* __restrict__ pooled, float* __restrict__ gcnt) {
  int i = blockIdx.x * blockDim.x + threadIdx.x;
  if (i >= NN) return;
  int g = batch[i];
  atomicAdd(&gcnt[g], 1.0f);
  const float* hr = h2 + (size_t)i * HIDC;
  float* pr = pooled + (size_t)g * HIDC;
#pragma unroll 8
  for (int c = 0; c < HIDC; ++c) {
    float v = hr[c] + b2[c];
    atomicAdd(&pr[c], (v > 0.0f) ? v : 0.0f);
  }
}

// ----------------------- final classifier -----------------------------
__global__ void final_kernel(const float* __restrict__ pooled,
                             const float* __restrict__ gcnt,
                             const float* __restrict__ Wc,
                             const float* __restrict__ bc,
                             float* __restrict__ out) {
  int t = threadIdx.x;
  if (t >= GG * COUT) return;
  int g = t / COUT, co = t % COUT;
  float inv = 1.0f / fmaxf(gcnt[g], 1.0f);
  float acc = bc[co];
#pragma unroll
  for (int c = 0; c < HIDC; ++c)
    acc = fmaf(pooled[(size_t)g * HIDC + c] * inv, Wc[c * COUT + co], acc);
  out[t] = acc;
}

// ======================================================================
extern "C" void kernel_launch(void* const* d_in, const int* in_sizes, int n_in,
                              void* d_out, int out_size, void* d_ws, size_t ws_size,
                              hipStream_t stream) {
  (void)in_sizes; (void)n_in; (void)out_size; (void)ws_size;
  const float* x     = (const float*)d_in[0];
  const int*   eidx  = (const int*)d_in[1];
  const int*   srcI  = eidx;
  const int*   dstI  = eidx + EE;
  const float* eattr = (const float*)d_in[2];
  const int*   batch = (const int*)d_in[3];
  const float* Wl1   = (const float*)d_in[4];
  const float* Wr1   = (const float*)d_in[5];
  const float* We1   = (const float*)d_in[6];
  const float* att1  = (const float*)d_in[7];
  const float* b1    = (const float*)d_in[8];
  const float* Wl2   = (const float*)d_in[9];
  const float* Wr2   = (const float*)d_in[10];
  const float* We2   = (const float*)d_in[11];
  const float* att2  = (const float*)d_in[12];
  const float* b2    = (const float*)d_in[13];
  const float* Wc    = (const float*)d_in[14];
  const float* bc    = (const float*)d_in[15];
  float* out = (float*)d_out;

  // ---- workspace carve-up (~115 MB), layer2 aliases dead L1 buffers ----
  char* ws = (char*)d_ws;
  auto carve = [&](size_t bytes) {
    char* p = ws; ws += (bytes + 255) & ~(size_t)255; return p;
  };
  float*     cnt       = (float*)    carve((size_t)NN * 4);
  float*     loop_attr = (float*)    carve((size_t)NN * EDIM * 4);
  float*     xl        = (float*)    carve((size_t)NN * HC1 * 4);
  float*     xr        = (float*)    carve((size_t)NN * HC1 * 4);
  float*     logits    = (float*)    carve((size_t)ETOT * NHEAD * 4);
  unsigned*  mx        = (unsigned*) carve((size_t)NN * NHEAD * 4);
  float*     den       = (float*)    carve((size_t)NN * NHEAD * 4);
  float*     h1        = (float*)    carve((size_t)NN * HC1 * 4);
  float*     h2        = (float*)    carve((size_t)NN * HIDC * 4);
  _Float16*  xh        = (_Float16*) carve((size_t)NN * HC1 * 2);  // f16 stage
  _Float16*  Wl1p      = (_Float16*) carve((size_t)FIN * HC1 * 2);
  _Float16*  Wr1p      = (_Float16*) carve((size_t)FIN * HC1 * 2);
  _Float16*  Wl2p      = (_Float16*) carve((size_t)HC1 * HIDC * 2);
  _Float16*  Wr2p      = (_Float16*) carve((size_t)HC1 * HIDC * 2);
  float*     pooled    = (float*)    carve((size_t)GG * HIDC * 4);
  float*     gcnt      = (float*)    carve((size_t)GG * 4);
  // layer-2 aliases (layer-1 contents dead by then)
  float*    xl2 = xl;  float* xr2 = xr;  float* logits2 = logits;
  unsigned* mx2 = mx;  float* den2 = den;
  _Float16* h1h = xh;   // xh dead after layer-1 GEMMs

  const int B = 256;
  auto blocks = [&](size_t n) { return (unsigned)((n + B - 1) / B); };

  // ---- deterministic per-call init (graph-replay safe) ----
  fill_u32_kernel<<<blocks((size_t)NN), B, 0, stream>>>((unsigned*)cnt, 0u, (size_t)NN);
  fill_u32_kernel<<<blocks((size_t)NN*EDIM), B, 0, stream>>>((unsigned*)loop_attr, 0u, (size_t)NN*EDIM);
  fill_u32_kernel<<<blocks((size_t)NN*NHEAD), B, 0, stream>>>(mx, FENC_NEG_INF, (size_t)NN*NHEAD);
  fill_u32_kernel<<<blocks((size_t)NN*NHEAD), B, 0, stream>>>((unsigned*)den, 0u, (size_t)NN*NHEAD);
  fill_u32_kernel<<<blocks((size_t)NN*HC1), B, 0, stream>>>((unsigned*)h1, 0u, (size_t)NN*HC1);
  fill_u32_kernel<<<blocks((size_t)NN*HIDC), B, 0, stream>>>((unsigned*)h2, 0u, (size_t)NN*HIDC);
  fill_u32_kernel<<<1, B, 0, stream>>>((unsigned*)pooled, 0u, (size_t)GG*HIDC);
  fill_u32_kernel<<<1, B, 0, stream>>>((unsigned*)gcnt, 0u, (size_t)GG);

  // ---- self-loop attrs: mean of incoming edge_attr per node ----
  degree_loop_accum_kernel<<<blocks((size_t)EE), B, 0, stream>>>(dstI, eattr, cnt, loop_attr);
  loop_div_kernel<<<blocks((size_t)NN*EDIM), B, 0, stream>>>(loop_attr, cnt);

  // ---- f16 staging: activations row-major, weights fragment-packed ----
  cvt_f16_kernel<<<blocks((size_t)NN*FIN), B, 0, stream>>>(x, xh, (size_t)NN*FIN);
  pack_w_kernel<FIN, HC1><<<blocks((size_t)FIN*HC1), B, 0, stream>>>(Wl1, Wl1p);
  pack_w_kernel<FIN, HC1><<<blocks((size_t)FIN*HC1), B, 0, stream>>>(Wr1, Wr1p);
  pack_w_kernel<HC1, HIDC><<<blocks((size_t)HC1*HIDC), B, 0, stream>>>(Wl2, Wl2p);
  pack_w_kernel<HC1, HIDC><<<blocks((size_t)HC1*HIDC), B, 0, stream>>>(Wr2, Wr2p);

  // ---- layer 1 dense projections via WMMA (3125 row tiles, 4/block) ----
  const unsigned gemm_blocks = (NN / 16 + 3) / 4;
  gemm_wmma_kernel<FIN, HC1><<<gemm_blocks, 128, 0, stream>>>(xh, Wl1p, xl);
  gemm_wmma_kernel<FIN, HC1><<<gemm_blocks, 128, 0, stream>>>(xh, Wr1p, xr);

  // ---- layer 1 attention ----
  edge_logits_kernel<NHEAD, HIDC, HC1><<<blocks((size_t)ETOT*NHEAD), B, 0, stream>>>(
      srcI, dstI, eattr, loop_attr, xl, xr, We1, att1, logits, mx);
  edge_expden_kernel<NHEAD><<<blocks((size_t)ETOT*NHEAD), B, 0, stream>>>(
      dstI, mx, logits, den);
  edge_aggregate_kernel<NHEAD, HIDC, HC1><<<blocks((size_t)ETOT*NHEAD), B, 0, stream>>>(
      srcI, dstI, logits, den, xl, h1);
  bias_relu_f16_kernel<<<blocks((size_t)NN*HC1), B, 0, stream>>>(
      h1, b1, h1h, HC1, (size_t)NN*HC1);

  // ---- layer 2 dense projections via WMMA ----
  gemm_wmma_kernel<HC1, HIDC><<<gemm_blocks, 128, 0, stream>>>(h1h, Wl2p, xl2);
  gemm_wmma_kernel<HC1, HIDC><<<gemm_blocks, 128, 0, stream>>>(h1h, Wr2p, xr2);

  // ---- layer 2 attention (heads=1) ----
  fill_u32_kernel<<<blocks((size_t)NN), B, 0, stream>>>(mx2, FENC_NEG_INF, (size_t)NN);
  fill_u32_kernel<<<blocks((size_t)NN), B, 0, stream>>>((unsigned*)den2, 0u, (size_t)NN);
  edge_logits_kernel<1, HIDC, HIDC><<<blocks((size_t)ETOT), B, 0, stream>>>(
      srcI, dstI, eattr, loop_attr, xl2, xr2, We2, att2, logits2, mx2);
  edge_expden_kernel<1><<<blocks((size_t)ETOT), B, 0, stream>>>(
      dstI, mx2, logits2, den2);
  edge_aggregate_kernel<1, HIDC, HIDC><<<blocks((size_t)ETOT), B, 0, stream>>>(
      srcI, dstI, logits2, den2, xl2, h2);

  // ---- bias+relu fused into global mean pool, then classifier ----
  pool_kernel<<<blocks((size_t)NN), B, 0, stream>>>(h2, batch, b2, pooled, gcnt);
  final_kernel<<<1, 128, 0, stream>>>(pooled, gcnt, Wc, bc, out);
}
